// Decoder_70033736728716
// MI455X (gfx1250) — matrix operand order repaired and checked by reading, not compiled, not run on hardware
//
#include <hip/hip_runtime.h>
#include <hip/hip_bf16.h>

typedef __attribute__((ext_vector_type(16))) _Float16 v16h;
typedef __attribute__((ext_vector_type(8)))  _Float16 v8h;
typedef __attribute__((ext_vector_type(8)))  float    v8f;

__device__ __forceinline__ v8f wmma16(v16h a, v16h b, v8f c) {
  return __builtin_amdgcn_wmma_f32_16x16x32_f16(false, a, false, b, (short)0, c, false, false);
}

__device__ __forceinline__ v8f bcast(float v) {
  return (v8f){v, v, v, v, v, v, v, v};
}

// A-fragment (16xK f16, MxK) from row-major [16][ld] at column kbase.
// Lane l: m = l&15, g = l>>4; elements 0..7 -> K = kbase+8g+i, 8..15 -> K = kbase+16+8g+i.
__device__ __forceinline__ v16h load_A(const _Float16* base, int ld, int kbase, int lane) {
  int m = lane & 15, g = lane >> 4;
  const _Float16* p = base + m * ld + kbase + 8 * g;
  union { v16h v; v8h h[2]; } u;
  u.h[0] = *(const v8h*)(p);
  u.h[1] = *(const v8h*)(p + 16);
  return u.v;
}

// B-fragment (KxN = 32x16) for D = A @ W.T : B[k][n] = W[n][k], W row-major [N][ld].
// Lane l: n = nbase + (l&15), g = l>>4; elements 0..7 -> K = kbase+8g+i, 8..15 -> K = kbase+16+8g+i.
__device__ __forceinline__ v16h load_B(const _Float16* W, int ld, int nbase, int kbase, int lane) {
  const _Float16* p = W + (size_t)(nbase + (lane & 15)) * ld + kbase + 8 * (lane >> 4);
  union { v16h v; v8h h[2]; } u;
  u.h[0] = *(const v8h*)(p);
  u.h[1] = *(const v8h*)(p + 16);
  return u.v;
}

__device__ __forceinline__ float fsig(float x) {
  return 1.0f / (1.0f + __expf(-x));
}
__device__ __forceinline__ float ftanh(float x) {
  x = fminf(fmaxf(x, -15.0f), 15.0f);
  float e = __expf(-2.0f * x);
  return (1.0f - e) / (1.0f + e);
}

// ---- weight prepack: f32 -> f16 (with padding for Wih K->32 and Wd N->32) ----
// ws layout (f16 elements): W1h[256*128] @0, W2h[256*256] @32768, Whh[768*256] @98304,
// WihP[768*32] @294912, WdP[32*256] @319488; total 327680 f16 = 640 KiB.
__global__ __launch_bounds__(256) void decoder_prepack(
    const float* __restrict__ W1, const float* __restrict__ W2,
    const float* __restrict__ Whh, const float* __restrict__ Wih,
    const float* __restrict__ Wd, _Float16* __restrict__ ws)
{
  int idx = blockIdx.x * 256 + threadIdx.x;
  if (idx < 32768) {
    ws[idx] = (_Float16)W1[idx];
  } else if (idx < 98304) {
    ws[idx] = (_Float16)W2[idx - 32768];
  } else if (idx < 294912) {
    ws[idx] = (_Float16)Whh[idx - 98304];
  } else if (idx < 319488) {
    int i = idx - 294912; int n = i >> 5, k = i & 31;
    ws[idx] = (_Float16)(k < 21 ? Wih[n * 21 + k] : 0.0f);
  } else if (idx < 327680) {
    int i = idx - 319488; int n = i >> 8, k = i & 255;
    ws[idx] = (_Float16)(n < 21 ? Wd[n * 256 + k] : 0.0f);
  }
}

// ---- main: one block = 16 batch rows, full encoder + 39 GRU steps ----
__global__ __launch_bounds__(256) void decoder_main(
    const float* __restrict__ latent,
    const float* __restrict__ b1, const float* __restrict__ b2,
    const float* __restrict__ bih, const float* __restrict__ bhh,
    const float* __restrict__ bd,
    const _Float16* __restrict__ W1h, const _Float16* __restrict__ W2h,
    const _Float16* __restrict__ WhhH, const _Float16* __restrict__ WihH,
    const _Float16* __restrict__ WdH,
    float* __restrict__ out)
{
  __shared__ __align__(16) _Float16 hF16[16 * 256];   // f16 h state (WMMA A operand)
  __shared__ __align__(16) _Float16 xF16[16 * 32];    // f16 x (pred), K padded 21->32
  __shared__ __align__(16) _Float16 latF[16 * 128];   // f16 latent tile
  __shared__ __align__(16) _Float16 tmpA[16 * 256];   // relu(mlp1) staging

  const int tid  = threadIdx.x;
  const int lane = tid & 31;
  const int wave = tid >> 5;
  const int row0 = blockIdx.x * 16;
  const int cn   = lane & 15;
  const int mb   = (lane >> 4) * 8;

  // ---- phase A: stage latent (f16), init x = start token, emit t=0 output ----
  for (int i = tid; i < 16 * 128; i += 256)
    latF[i] = (_Float16)latent[(size_t)(row0 + (i >> 7)) * 128 + (i & 127)];
  for (int i = tid; i < 16 * 32; i += 256) {
    int k = i & 31;
    xF16[i] = (_Float16)(k == 0 ? 32.0f : (k < 21 ? -32.0f : 0.0f));
  }
  for (int i = tid; i < 16 * 21; i += 256) {
    int m = i / 21, a = i - m * 21;
    out[(size_t)(row0 + m) * 840 + a] = (a == 0) ? 32.0f : -32.0f;
  }
  __syncthreads();

  // ---- GEMM1: relu(latent @ W1.T + b1) -> tmpA (f16). 16 N-tiles, 2 per wave, K=128 ----
  {
    int nb0 = wave * 32, nb1 = nb0 + 16;
    v8f c0 = bcast(b1[nb0 + cn]);
    v8f c1 = bcast(b1[nb1 + cn]);
#pragma unroll
    for (int k = 0; k < 4; ++k) {
      v16h a = load_A(latF, 128, k * 32, lane);
      c0 = wmma16(a, load_B(W1h, 128, nb0, k * 32, lane), c0);
      c1 = wmma16(a, load_B(W1h, 128, nb1, k * 32, lane), c1);
    }
#pragma unroll
    for (int q = 0; q < 8; ++q) {
      tmpA[(mb + q) * 256 + nb0 + cn] = (_Float16)fmaxf(c0[q], 0.0f);
      tmpA[(mb + q) * 256 + nb1 + cn] = (_Float16)fmaxf(c1[q], 0.0f);
    }
  }
  __syncthreads();

  // ---- GEMM2: hidden = tmpA @ W2.T + b2 -> h (f32 regs + f16 LDS), K=256 ----
  v8f hreg0, hreg1;
  {
    int nb0 = wave * 32, nb1 = nb0 + 16;
    v8f c0 = bcast(b2[nb0 + cn]);
    v8f c1 = bcast(b2[nb1 + cn]);
#pragma unroll
    for (int k = 0; k < 8; ++k) {
      v16h a = load_A(tmpA, 256, k * 32, lane);
      c0 = wmma16(a, load_B(W2h, 256, nb0, k * 32, lane), c0);
      c1 = wmma16(a, load_B(W2h, 256, nb1, k * 32, lane), c1);
    }
    hreg0 = c0; hreg1 = c1;
#pragma unroll
    for (int q = 0; q < 8; ++q) {
      hF16[(mb + q) * 256 + nb0 + cn] = (_Float16)c0[q];
      hF16[(mb + q) * 256 + nb1 + cn] = (_Float16)c1[q];
    }
  }
  __syncthreads();

  // Wave w owns gate columns [32w, 32w+32) of each of the r/z/n sections.
  const int rb0 = wave * 32, rb1 = rb0 + 16;
  const float bR0 = bih[rb0 + cn]       + bhh[rb0 + cn];
  const float bR1 = bih[rb1 + cn]       + bhh[rb1 + cn];
  const float bZ0 = bih[256 + rb0 + cn] + bhh[256 + rb0 + cn];
  const float bZ1 = bih[256 + rb1 + cn] + bhh[256 + rb1 + cn];
  const float bNI0 = bih[512 + rb0 + cn];
  const float bNI1 = bih[512 + rb1 + cn];
  const float bNH0 = bhh[512 + rb0 + cn];
  const float bNH1 = bhh[512 + rb1 + cn];
  const float bD   = (wave * 16 + cn) < 21 ? bd[wave * 16 + cn] : 0.0f;

#pragma unroll 1
  for (int t = 1; t < 40; ++t) {
    // hoist A-fragments of h and x (read LDS once per step)
    v16h ah[8];
#pragma unroll
    for (int k = 0; k < 8; ++k) ah[k] = load_A(hF16, 256, k * 32, lane);
    v16h xa = load_A(xF16, 32, 0, lane);
    __syncthreads();  // all reads of hF16/xF16 done before anyone rewrites them

    v8f aR0 = bcast(bR0), aR1 = bcast(bR1);
    v8f aZ0 = bcast(bZ0), aZ1 = bcast(bZ1);
    v8f aNI0 = bcast(bNI0), aNI1 = bcast(bNI1);
    v8f aNH0 = bcast(bNH0), aNH1 = bcast(bNH1);

    // input-gate GEMM (K padded to 32): one WMMA per tile
    aR0  = wmma16(xa, load_B(WihH, 32, rb0,        0, lane), aR0);
    aR1  = wmma16(xa, load_B(WihH, 32, rb1,        0, lane), aR1);
    aZ0  = wmma16(xa, load_B(WihH, 32, 256 + rb0,  0, lane), aZ0);
    aZ1  = wmma16(xa, load_B(WihH, 32, 256 + rb1,  0, lane), aZ1);
    aNI0 = wmma16(xa, load_B(WihH, 32, 512 + rb0,  0, lane), aNI0);
    aNI1 = wmma16(xa, load_B(WihH, 32, 512 + rb1,  0, lane), aNI1);

    // hidden-gate GEMM: K=256, six independent accumulator chains
#pragma unroll
    for (int k = 0; k < 8; ++k) {
      int kb = k * 32;
      v16h a = ah[k];
      aR0  = wmma16(a, load_B(WhhH, 256, rb0,       kb, lane), aR0);
      aR1  = wmma16(a, load_B(WhhH, 256, rb1,       kb, lane), aR1);
      aZ0  = wmma16(a, load_B(WhhH, 256, 256 + rb0, kb, lane), aZ0);
      aZ1  = wmma16(a, load_B(WhhH, 256, 256 + rb1, kb, lane), aZ1);
      aNH0 = wmma16(a, load_B(WhhH, 256, 512 + rb0, kb, lane), aNH0);
      aNH1 = wmma16(a, load_B(WhhH, 256, 512 + rb1, kb, lane), aNH1);
    }

    // GRU gates entirely in registers (fragment layouts line up across sections)
#pragma unroll
    for (int q = 0; q < 8; ++q) {
      float r0 = fsig(aR0[q]);
      float z0 = fsig(aZ0[q]);
      float n0 = ftanh(aNI0[q] + r0 * aNH0[q]);
      float h0 = (1.0f - z0) * n0 + z0 * hreg0[q];
      hreg0[q] = h0;
      hF16[(mb + q) * 256 + rb0 + cn] = (_Float16)h0;

      float r1 = fsig(aR1[q]);
      float z1 = fsig(aZ1[q]);
      float n1 = ftanh(aNI1[q] + r1 * aNH1[q]);
      float h1 = (1.0f - z1) * n1 + z1 * hreg1[q];
      hreg1[q] = h1;
      hF16[(mb + q) * 256 + rb1 + cn] = (_Float16)h1;
    }
    __syncthreads();  // new hF16 visible for pred

    // pred = h_new @ Wd.T + bd  (N=21 padded to 32 -> 2 N-tiles on waves 0,1)
    if (wave < 2) {
      int nb  = wave * 16;
      int col = nb + cn;
      v8f c = bcast(bD);
#pragma unroll
      for (int k = 0; k < 8; ++k)
        c = wmma16(load_A(hF16, 256, k * 32, lane),
                   load_B(WdH, 256, nb, k * 32, lane), c);
      if (col < 21) {
#pragma unroll
        for (int q = 0; q < 8; ++q) {
          int m = mb + q;
          out[(size_t)(row0 + m) * 840 + t * 21 + col] = c[q];
          xF16[m * 32 + col] = (_Float16)c[q];   // becomes next step's input
        }
      }
    }
    __syncthreads();  // xF16 ready for next step
  }
}

extern "C" void kernel_launch(void* const* d_in, const int* in_sizes, int n_in,
                              void* d_out, int out_size, void* d_ws, size_t ws_size,
                              hipStream_t stream) {
  const float* latent = (const float*)d_in[0];
  const float* W1  = (const float*)d_in[1];
  const float* b1  = (const float*)d_in[2];
  const float* W2  = (const float*)d_in[3];
  const float* b2  = (const float*)d_in[4];
  const float* Wih = (const float*)d_in[5];
  const float* Whh = (const float*)d_in[6];
  const float* bih = (const float*)d_in[7];
  const float* bhh = (const float*)d_in[8];
  const float* Wd  = (const float*)d_in[9];
  const float* bd  = (const float*)d_in[10];
  float* out = (float*)d_out;

  _Float16* ws   = (_Float16*)d_ws;
  _Float16* W1h  = ws;
  _Float16* W2h  = ws + 32768;
  _Float16* WhhH = ws + 98304;
  _Float16* WihH = ws + 294912;
  _Float16* WdH  = ws + 319488;

  decoder_prepack<<<1280, 256, 0, stream>>>(W1, W2, Whh, Wih, Wd, ws);
  decoder_main<<<1024, 256, 0, stream>>>(latent, b1, b2, bih, bhh, bd,
                                         W1h, W2h, WhhH, WihH, WdH, out);
}
